// MultiHeadAttention_31147102830894
// MI455X (gfx1250) — matrix operand order repaired and checked by reading, not compiled, not run on hardware
//
#include <hip/hip_runtime.h>
#include <cstddef>
#include <type_traits>

// ---------------------------------------------------------------------------
// Problem constants (from the reference)
// ---------------------------------------------------------------------------
#define D_MODEL  768
#define HEADS    6
#define HEAD_DIM 128           // D_MODEL / HEADS
#define NTOK     (8 * 2048)    // B * S = 16384 tokens

typedef __attribute__((ext_vector_type(16))) _Float16 v16h;
typedef __attribute__((ext_vector_type(8)))  _Float16 v8h;
typedef __attribute__((ext_vector_type(8)))  float    v8f;

static __device__ __forceinline__ v8f wmma_f16(v16h a, v16h b, v8f c) {
  // D = A(16x32 f16) * B(32x16 f16) + C(16x16 f32)
  return __builtin_amdgcn_wmma_f32_16x16x32_f16(
      /*neg_a=*/false, a, /*neg_b=*/false, b,
      /*c_mod=*/(short)0, c, /*reuse_a=*/false, /*reuse_b=*/false);
}

// ---------------------------------------------------------------------------
// GEMM: Y(MxN) = X(MxK) @ W(NxK)^T (+ bias), f32 accumulate via WMMA f16.
// Block = 256 threads (8 waves). Each wave computes a 16x64 tile
// (one A fragment reused across 4 B fragments). Grid: (N/64, M/128).
// All tile loads are 128-bit vector loads (float4 / v8h).
//
// MODE: 0 = f16 row-major store (v intermediate)
//       1 = f16 per-token transposed store: y[row, (n%128)*6 + n/128]
//           (q/k intermediates -> contiguous head-vectors for attention)
//       2 = f32 row-major store + bias (final output)
// ---------------------------------------------------------------------------
template <typename XT, int MODE>
__global__ __launch_bounds__(256)
void gemm_xwT(const XT* __restrict__ X, const float* __restrict__ W,
              const float* __restrict__ bias, void* __restrict__ Yv,
              int M, int N, int K) {
  const int lane  = threadIdx.x & 31;
  const int wave  = threadIdx.x >> 5;
  const int lm    = lane & 15;
  const int half  = lane >> 4;          // 0 or 1
  const int mrow  = blockIdx.y * 128 + wave * 16;
  const int ncol  = blockIdx.x * 64;

  v8f acc[4] = {};

  const XT* xrow = X + (size_t)(mrow + lm) * K;
  const int kh = half * 8;              // A-fragment K sub-offset for this lane

  for (int k0 = 0; k0 < K; k0 += 32) {
    // A fragment: row (mrow+lm), elems 0-7 <- K=k0+kh.., elems 8-15 <- K=k0+16+kh..
    v16h a;
    if constexpr (std::is_same<XT, float>::value) {
      const float* xa = xrow + k0 + kh;
      float fa[16];
      *(float4*)(fa + 0)  = *(const float4*)(xa + 0);
      *(float4*)(fa + 4)  = *(const float4*)(xa + 4);
      *(float4*)(fa + 8)  = *(const float4*)(xa + 16);
      *(float4*)(fa + 12) = *(const float4*)(xa + 20);
#pragma unroll
      for (int e = 0; e < 16; ++e) a[e] = (_Float16)fa[e];
    } else {
      const _Float16* xa = xrow + k0 + kh;
      v8h lo = *(const v8h*)(xa);
      v8h hi = *(const v8h*)(xa + 16);
#pragma unroll
      for (int e = 0; e < 8; ++e) { a[e] = lo[e]; a[8 + e] = hi[e]; }
    }

#pragma unroll
    for (int j = 0; j < 4; ++j) {
      // B fragment: column n = ncol + j*16 + lm, elems 0-7 <- K=k0+half*8.., elems 8-15 <- K=k0+16+half*8..
      const float* wrow = W + (size_t)(ncol + j * 16 + lm) * K + k0 + half * 8;
      float fb[16];
      *(float4*)(fb + 0)  = *(const float4*)(wrow + 0);
      *(float4*)(fb + 4)  = *(const float4*)(wrow + 4);
      *(float4*)(fb + 8)  = *(const float4*)(wrow + 16);
      *(float4*)(fb + 12) = *(const float4*)(wrow + 20);
      v16h b;
#pragma unroll
      for (int e = 0; e < 16; ++e) b[e] = (_Float16)fb[e];
      acc[j] = wmma_f16(a, b, acc[j]);
    }
  }

  // C/D layout: vgpr r -> row r + 8*half, col = lm (within a 16x16 tile)
#pragma unroll
  for (int j = 0; j < 4; ++j) {
    const int col = ncol + j * 16 + lm;
#pragma unroll
    for (int r = 0; r < 8; ++r) {
      const int row = mrow + r + (half << 3);
      const float val = acc[j][r];
      if constexpr (MODE == 2) {
        ((float*)Yv)[(size_t)row * N + col] = val + bias[col];
      } else if constexpr (MODE == 1) {
        // per-token transpose: n = h*128 + d  ->  store at d*HEADS + h
        ((_Float16*)Yv)[(size_t)row * N + (col & (HEAD_DIM - 1)) * HEADS + (col >> 7)] =
            (_Float16)val;
      } else {
        ((_Float16*)Yv)[(size_t)row * N + col] = (_Float16)val;
      }
    }
  }
}

// ---------------------------------------------------------------------------
// Per-token attention over head_dim:
//   scores(128x128) = q_mat(128x6) @ k_mat(128x6)^T / scale
//   probs = softmax(scores, axis=-1)
//   att(128x6) = probs @ v_mat(128x6)
// qT/kT are per-token transposed: qT[t, d*6+h] -> contiguous head vectors.
// v stays row-major: vlin[t, h*128+e] -> contiguous over e for B fragments.
// One block (8 waves) per token; wave w owns d-rows [w*16, w*16+16).
// Softmax rides the C-layout (each score row on one 16-lane half,
// __shfl_xor 1/2/4/8 reductions); probs restaged through per-wave LDS tiles
// to build A fragments for the second WMMA.
// ---------------------------------------------------------------------------
__global__ __launch_bounds__(256)
void attn_tok(const _Float16* __restrict__ qT, const _Float16* __restrict__ kT,
              const _Float16* __restrict__ vlin, const float* __restrict__ unc,
              _Float16* __restrict__ merged) {
  __shared__ _Float16 lds_p[8][16][HEAD_DIM];   // 32 KB, per-wave private tiles

  const int t    = blockIdx.x;
  const int lane = threadIdx.x & 31;
  const int wave = threadIdx.x >> 5;
  const int lm   = lane & 15;
  const int half = lane >> 4;

  float us = 0.0f;
#pragma unroll
  for (int h = 0; h < HEADS; ++h) us += unc[h];
  const float inv_scale = 1.0f / (sqrtf((float)HEAD_DIM) * us);

  const _Float16* qt = qT   + (size_t)t * D_MODEL;
  const _Float16* kt = kT   + (size_t)t * D_MODEL;
  const _Float16* vt = vlin + (size_t)t * D_MODEL;

  // A fragment of q: only lanes 0-15, K(h) = 0..5 valid; contiguous 6 f16.
  v16h aq = {};
  if (half == 0) {
    const _Float16* qp = qt + (wave * 16 + lm) * HEADS;
#pragma unroll
    for (int h = 0; h < HEADS; ++h) aq[h] = qp[h];
  }

  // scores: 8 e-tiles of 16x16
  v8f acc[8] = {};
#pragma unroll
  for (int j = 0; j < 8; ++j) {
    v16h bk = {};
    if (half == 0) {                      // lanes 0-15 carry K(h)=0..5 of B
      const _Float16* kp = kt + (j * 16 + lm) * HEADS;
#pragma unroll
      for (int h = 0; h < HEADS; ++h) bk[h] = kp[h];
    }
    acc[j] = wmma_f16(aq, bk, acc[j]);
  }

  // Row-wise softmax over the 128 e-columns (8 tiles x 16 lanes per half).
#pragma unroll
  for (int r = 0; r < 8; ++r) {
    float mx = -3.402823466e38f;
#pragma unroll
    for (int j = 0; j < 8; ++j) {
      acc[j][r] *= inv_scale;
      mx = fmaxf(mx, acc[j][r]);
    }
#pragma unroll
    for (int off = 1; off < 16; off <<= 1) mx = fmaxf(mx, __shfl_xor(mx, off, 32));
    float sum = 0.0f;
#pragma unroll
    for (int j = 0; j < 8; ++j) {
      const float p = __expf(acc[j][r] - mx);
      acc[j][r] = p;
      sum += p;
    }
#pragma unroll
    for (int off = 1; off < 16; off <<= 1) sum += __shfl_xor(sum, off, 32);
    const float rs = 1.0f / sum;
    const int row = r + (half << 3);
#pragma unroll
    for (int j = 0; j < 8; ++j)
      lds_p[wave][row][j * 16 + lm] = (_Float16)(acc[j][r] * rs);
  }
  // Same-wave LDS RAW: DS ops are in-order per wave; compiler inserts dscnt waits.

  // att = probs(16x128) @ v(128x6): 4 K-steps of 32.
  v8f oacc = {};
#pragma unroll
  for (int kb = 0; kb < HEAD_DIM; kb += 32) {
    const int kh = half * 8;
    const _Float16* lp = &lds_p[wave][lm][kb + kh];
    v8h p0 = *(const v8h*)(lp);
    v8h p1 = *(const v8h*)(lp + 16);
    v16h pa;
#pragma unroll
    for (int e = 0; e < 8; ++e) { pa[e] = p0[e]; pa[8 + e] = p1[e]; }

    v16h bv = {};
    if (lm < HEADS) {                     // only 6 valid output columns (heads)
      const _Float16* vp = vt + lm * HEAD_DIM + kb + half * 8;
      v8h b0 = *(const v8h*)(vp);
      v8h b1 = *(const v8h*)(vp + 16);
#pragma unroll
      for (int e = 0; e < 8; ++e) { bv[e] = b0[e]; bv[8 + e] = b1[e]; }
    }
    oacc = wmma_f16(pa, bv, oacc);
  }

  // combine_heads: merged[t, h*128 + d] = att[d, h]
  if (lm < HEADS) {
#pragma unroll
    for (int r = 0; r < 8; ++r) {
      const int d = wave * 16 + r + (half << 3);
      merged[(size_t)t * D_MODEL + lm * HEAD_DIM + d] = (_Float16)oacc[r];
    }
  }
}

// ---------------------------------------------------------------------------
// Launcher: 3 projection GEMMs -> f16 ws (q/k transposed per token),
// per-token attention -> f16 ws, output projection (+bias) -> f32 d_out.
// Workspace: 4 x 16384x768 f16 = 96 MB.
// ---------------------------------------------------------------------------
extern "C" void kernel_launch(void* const* d_in, const int* in_sizes, int n_in,
                              void* d_out, int out_size, void* d_ws, size_t ws_size,
                              hipStream_t stream) {
  const float* Q   = (const float*)d_in[0];
  const float* Kin = (const float*)d_in[1];
  const float* V   = (const float*)d_in[2];
  const float* unc = (const float*)d_in[3];
  const float* Wq  = (const float*)d_in[4];
  const float* Wk  = (const float*)d_in[5];
  const float* Wv  = (const float*)d_in[6];
  const float* Wo  = (const float*)d_in[7];
  const float* bo  = (const float*)d_in[8];
  float* out = (float*)d_out;

  const size_t elems = (size_t)NTOK * D_MODEL;
  _Float16* qh = (_Float16*)d_ws;
  _Float16* kh = qh + elems;
  _Float16* vh = kh + elems;
  _Float16* mh = vh + elems;

  dim3 grid(D_MODEL / 64, NTOK / 128);   // (12, 128)
  dim3 blk(256);

  gemm_xwT<float, 1><<<grid, blk, 0, stream>>>(Q,   Wq, nullptr, qh, NTOK, D_MODEL, D_MODEL);
  gemm_xwT<float, 1><<<grid, blk, 0, stream>>>(Kin, Wk, nullptr, kh, NTOK, D_MODEL, D_MODEL);
  gemm_xwT<float, 0><<<grid, blk, 0, stream>>>(V,   Wv, nullptr, vh, NTOK, D_MODEL, D_MODEL);

  attn_tok<<<dim3(NTOK), blk, 0, stream>>>(qh, kh, vh, unc, mh);

  gemm_xwT<_Float16, 2><<<grid, blk, 0, stream>>>(mh, Wo, bo, out, NTOK, D_MODEL, D_MODEL);
}
